// MoEFusion_48473000902765
// MI455X (gfx1250) — matrix operand order repaired and checked by reading, not compiled, not run on hardware
//
#include <hip/hip_runtime.h>
#include <math.h>

// ---------------- problem constants ----------------
#define BB       16384
#define IN_DIM   1024
#define OUT_DIM  1024
#define NE       8
#define LN_EPS   1e-5f

// ---------------- tiling ----------------
#define ROWS     64                 // rows per workgroup (4 M-tiles)
#define MT       4                  // m-tiles per wave
#define WAVES    16
#define THREADS  (WAVES * 32)       // 512
#define KT       (IN_DIM / 32)      // 32 k-steps of 32
#define NT_TOT   (OUT_DIM / 16)     // 64 n-tiles
#define NT_W     4                  // n-tiles per wave (16 waves * 4 = 64)
#define HSTR     (OUT_DIM + 8)      // LDS row stride (bf16): +16B pad -> 4-bank rotate/row

typedef unsigned short u16;
typedef __attribute__((ext_vector_type(8)))  float   v8f;
typedef __attribute__((ext_vector_type(8)))  __bf16  v8bf;
typedef __attribute__((ext_vector_type(16))) __bf16  v16bf;

__device__ __forceinline__ u16 f2bf(float f) {
    unsigned u = __float_as_uint(f);
    u += 0x7FFFu + ((u >> 16) & 1u);        // round-to-nearest-even
    return (u16)(u >> 16);
}

// Build a 16x32 bf16 A-fragment (8 VGPRs) from two 16B chunks.
// ISA layout: lanes 0-15 hold K 0-7 (v0-3) and K 16-23 (v4-7); lanes 16-31 hold K 8-15 / 24-31.
template <typename P>
__device__ __forceinline__ v16bf frag_a(P p_lo, P p_hi) {
    v8bf lo = *reinterpret_cast<const v8bf*>(p_lo);
    v8bf hi = *reinterpret_cast<const v8bf*>(p_hi);
    v16bf f;
#pragma unroll
    for (int i = 0; i < 8; ++i) { f[i] = lo[i]; f[i + 8] = hi[i]; }
    return f;
}

// ---------------- prep: combined activations -> bf16 row-major ----------------
__global__ void prep_cmb(const float* __restrict__ zs, const float* __restrict__ ze,
                         u16* __restrict__ cmb) {
    size_t idx = (size_t)blockIdx.x * blockDim.x + threadIdx.x;  // over B*1024
    int i = (int)(idx & 1023);
    size_t b = idx >> 10;
    float v = (i < 512) ? zs[b * 512 + i] : ze[b * 512 + (i - 512)];
    cmb[idx] = f2bf(v);
}

// ---------------- prep: weights fp32 [E][K][N] -> bf16 swizzled B-fragment blocks ----
// dst layout: [E][kt][nt][lane 0..31][kk 0..15], lane = ((kl/8)%2)*16 + n%16, kk = kl%8 + (kl/16)*8, so a wave's
// B fragment is 16 contiguous bf16 per lane (two global_load_b128).
__global__ void prep_wswz(const float* __restrict__ w, u16* __restrict__ dst) {
    size_t idx = (size_t)blockIdx.x * blockDim.x + threadIdx.x;  // over E*1024*1024
    int n = (int)(idx & 1023);
    int k = (int)((idx >> 10) & 1023);
    int e = (int)(idx >> 20);
    int kt = k >> 5, kl = k & 31, nt = n >> 4, nl = n & 15;
    int lane = (((kl >> 3) & 1) << 4) | nl;
    int kk   = (kl & 7) | ((kl >> 4) << 3);
    size_t doff = ((((size_t)e * KT + kt) * NT_TOT + nt) * 32 + lane) * 16 + kk;
    dst[doff] = f2bf(w[idx]);
}

// ---------------- router: logits + softmax, one wave per row ----------------
__global__ void router_kernel(const float* __restrict__ zs, const float* __restrict__ ze,
                              const float* __restrict__ rw, const float* __restrict__ rb,
                              float* __restrict__ wgt) {
    int wave = threadIdx.x >> 5, lane = threadIdx.x & 31;
    int row = blockIdx.x * 8 + wave;
    float acc[NE];
#pragma unroll
    for (int e = 0; e < NE; ++e) acc[e] = 0.f;
    for (int i = lane; i < IN_DIM; i += 32) {
        float c = (i < 512) ? zs[(size_t)row * 512 + i] : ze[(size_t)row * 512 + (i - 512)];
        const float* wr = rw + (size_t)i * NE;
#pragma unroll
        for (int e = 0; e < NE; ++e) acc[e] = fmaf(c, wr[e], acc[e]);
    }
#pragma unroll
    for (int e = 0; e < NE; ++e) {
        float v = acc[e];
#pragma unroll
        for (int m = 16; m >= 1; m >>= 1) v += __shfl_xor(v, m, 32);
        acc[e] = v + rb[e];
    }
    if (lane == 0) {
        float mx = acc[0];
#pragma unroll
        for (int e = 1; e < NE; ++e) mx = fmaxf(mx, acc[e]);
        float s = 0.f, ex[NE];
#pragma unroll
        for (int e = 0; e < NE; ++e) { ex[e] = __expf(acc[e] - mx); s += ex[e]; }
        float inv = 1.f / s;
#pragma unroll
        for (int e = 0; e < NE; ++e) wgt[(size_t)row * NE + e] = ex[e] * inv;
    }
}

// ---------------- fused MoE main kernel ----------------
// out must be pre-zeroed; per-expert normalized contributions accumulate with f32 atomics.
__global__ __launch_bounds__(THREADS)
void moe_main(const u16* __restrict__ cmb, const u16* __restrict__ w1s,
              const u16* __restrict__ w2s, const float* __restrict__ b1,
              const float* __restrict__ b2, const float* __restrict__ gamma,
              const float* __restrict__ beta, const float* __restrict__ wgt,
              float* __restrict__ out) {
    __shared__ __attribute__((aligned(16))) u16 hS[ROWS * HSTR];  // ~129 KB
    __shared__ float red[ROWS][2];

    const int tid  = threadIdx.x;
    const int wave = tid >> 5, lane = tid & 31;
    const int half = lane >> 4, lh = lane & 15;
    const int row0 = blockIdx.x * ROWS;
    const int nt0  = wave * NT_W;

    for (int e = 0; e < NE; ++e) {
        __syncthreads();                       // hS / red safe to rewrite
        if (tid < ROWS) { red[tid][0] = 0.f; red[tid][1] = 0.f; }

        float b1c[NT_W], b2c[NT_W], gc[NT_W], bc[NT_W];
#pragma unroll
        for (int ni = 0; ni < NT_W; ++ni) {
            int col = (nt0 + ni) * 16 + lh;
            b1c[ni] = b1[e * OUT_DIM + col];
            b2c[ni] = b2[e * OUT_DIM + col];
            gc[ni]  = gamma[e * OUT_DIM + col];
            bc[ni]  = beta[e * OUT_DIM + col];
        }

        // -------- stage A: h = gelu(cmb @ w1[e] + b1[e]) --------
        v8f acc[MT][NT_W];
#pragma unroll
        for (int mi = 0; mi < MT; ++mi)
#pragma unroll
            for (int ni = 0; ni < NT_W; ++ni) acc[mi][ni] = v8f{};

        const u16* w1e = w1s + (size_t)e * KT * NT_TOT * 512;
        for (int kt = 0; kt < KT; ++kt) {
            v16bf af[MT];
#pragma unroll
            for (int mi = 0; mi < MT; ++mi) {
                const u16* p = cmb + (size_t)(row0 + mi * 16 + lh) * IN_DIM + kt * 32 + half * 8;
                af[mi] = frag_a(p, p + 16);
            }
#pragma unroll
            for (int ni = 0; ni < NT_W; ++ni) {
                const u16* bp = w1e + ((size_t)(kt * NT_TOT + nt0 + ni) * 32 + lane) * 16;
                v16bf bf = *reinterpret_cast<const v16bf*>(bp);
#pragma unroll
                for (int mi = 0; mi < MT; ++mi)
                    acc[mi][ni] = __builtin_amdgcn_wmma_f32_16x16x32_bf16(
                        false, af[mi], false, bf, (short)0, acc[mi][ni], false, false);
            }
        }
        // epilogue A: bias + exact gelu -> LDS (bf16, padded row-major)
#pragma unroll
        for (int mi = 0; mi < MT; ++mi)
#pragma unroll
            for (int ni = 0; ni < NT_W; ++ni)
#pragma unroll
                for (int r = 0; r < 8; ++r) {
                    int rl  = mi * 16 + half * 8 + r;
                    int col = (nt0 + ni) * 16 + lh;
                    float x = acc[mi][ni][r] + b1c[ni];
                    float g = 0.5f * x * (1.f + erff(x * 0.70710678118654752f));
                    hS[rl * HSTR + col] = f2bf(g);
                }
        __syncthreads();

        // -------- stage B: y = h @ w2[e] + b2[e] --------
#pragma unroll
        for (int mi = 0; mi < MT; ++mi)
#pragma unroll
            for (int ni = 0; ni < NT_W; ++ni) acc[mi][ni] = v8f{};

        const u16* w2e = w2s + (size_t)e * KT * NT_TOT * 512;
        for (int kt = 0; kt < KT; ++kt) {
            v16bf af[MT];
#pragma unroll
            for (int mi = 0; mi < MT; ++mi) {
                const u16* p = hS + (mi * 16 + lh) * HSTR + kt * 32 + half * 8;
                af[mi] = frag_a(p, p + 16);
            }
#pragma unroll
            for (int ni = 0; ni < NT_W; ++ni) {
                const u16* bp = w2e + ((size_t)(kt * NT_TOT + nt0 + ni) * 32 + lane) * 16;
                v16bf bf = *reinterpret_cast<const v16bf*>(bp);
#pragma unroll
                for (int mi = 0; mi < MT; ++mi)
                    acc[mi][ni] = __builtin_amdgcn_wmma_f32_16x16x32_bf16(
                        false, af[mi], false, bf, (short)0, acc[mi][ni], false, false);
            }
        }
        // epilogue B: bias + LayerNorm stats (half-wave shuffle reduce + LDS atomics)
#pragma unroll
        for (int mi = 0; mi < MT; ++mi)
#pragma unroll
            for (int r = 0; r < 8; ++r) {
                float s = 0.f, s2 = 0.f;
#pragma unroll
                for (int ni = 0; ni < NT_W; ++ni) {
                    float v = acc[mi][ni][r] + b2c[ni];
                    acc[mi][ni][r] = v;
                    s += v; s2 += v * v;
                }
#pragma unroll
                for (int m = 1; m < 16; m <<= 1) {
                    s  += __shfl_xor(s,  m, 32);
                    s2 += __shfl_xor(s2, m, 32);
                }
                if (lh == 0) {
                    int rl = mi * 16 + half * 8 + r;
                    atomicAdd(&red[rl][0], s);
                    atomicAdd(&red[rl][1], s2);
                }
            }
        __syncthreads();

        // normalize + softmax-weighted accumulate straight into out (f32 atomics)
#pragma unroll
        for (int mi = 0; mi < MT; ++mi)
#pragma unroll
            for (int r = 0; r < 8; ++r) {
                int rl = mi * 16 + half * 8 + r;
                float mu   = red[rl][0] * (1.f / OUT_DIM);
                float var  = red[rl][1] * (1.f / OUT_DIM) - mu * mu;
                float rstd = rsqrtf(var + LN_EPS);
                float wb   = wgt[(size_t)(row0 + rl) * NE + e];
                float ws_  = wb * rstd;
#pragma unroll
                for (int ni = 0; ni < NT_W; ++ni) {
                    int col = (nt0 + ni) * 16 + lh;
                    float contrib = ws_ * (acc[mi][ni][r] - mu) * gc[ni] + wb * bc[ni];
                    atomicAdd(&out[(size_t)(row0 + rl) * OUT_DIM + col], contrib);
                }
            }
    }
}

// ---------------- launch ----------------
extern "C" void kernel_launch(void* const* d_in, const int* in_sizes, int n_in,
                              void* d_out, int out_size, void* d_ws, size_t ws_size,
                              hipStream_t stream) {
    const float* z_s      = (const float*)d_in[0];
    const float* z_e      = (const float*)d_in[1];
    const float* router_w = (const float*)d_in[2];
    const float* router_b = (const float*)d_in[3];
    const float* w1       = (const float*)d_in[4];
    const float* b1       = (const float*)d_in[5];
    const float* w2       = (const float*)d_in[6];
    const float* b2       = (const float*)d_in[7];
    const float* gamma    = (const float*)d_in[8];
    const float* beta     = (const float*)d_in[9];
    float* out = (float*)d_out;

    // ws layout: router weights (512KB) | cmb bf16 (32MB) | w1 bf16 swz (16MB) | w2 bf16 swz (16MB)
    char* ws = (char*)d_ws;
    float* wgt = (float*)ws;
    u16* cmb = (u16*)(ws + (512ull << 10));
    u16* w1s = (u16*)(ws + (512ull << 10) + (32ull << 20));
    u16* w2s = (u16*)(ws + (512ull << 10) + (48ull << 20));

    hipMemsetAsync(out, 0, (size_t)out_size * sizeof(float), stream);
    prep_cmb<<<(BB * IN_DIM) / 256, 256, 0, stream>>>(z_s, z_e, cmb);
    prep_wswz<<<(NE * IN_DIM * OUT_DIM) / 256, 256, 0, stream>>>(w1, w1s);
    prep_wswz<<<(NE * OUT_DIM * OUT_DIM) / 256, 256, 0, stream>>>(w2, w2s);
    router_kernel<<<BB / 8, 256, 0, stream>>>(z_s, z_e, router_w, router_b, wgt);
    moe_main<<<BB / ROWS, THREADS, 0, stream>>>(cmb, w1s, w2s, b1, b2, gamma, beta, wgt, out);
}
